// E61bAdditiveDecayCell_4887672783106
// MI455X (gfx1250) — compile-verified, matrix operand order checked by reading, not run on hardware
//
#include <hip/hip_runtime.h>

#define T_DIM 2048
#define B_DIM 8
#define D_DIM 1024
#define M_DIM (T_DIM * B_DIM)      // 16384 rows of the GEMMs
#define BD    (B_DIM * D_DIM)      // 8192 independent recurrence lanes
#define NCHUNK 64
#define CHUNKLEN (T_DIM / NCHUNK)  // 32

#define KSTEP 64
#define LDA 72                     // padded LDS row stride (bank-conflict relief)

typedef __attribute__((ext_vector_type(16))) __bf16          v16bf;
typedef __attribute__((ext_vector_type(8)))  float           v8f;
typedef __attribute__((ext_vector_type(4)))  unsigned short  v4us;
typedef __attribute__((ext_vector_type(8)))  unsigned short  v8us;
typedef __attribute__((ext_vector_type(16))) unsigned short  v16us;

static __device__ __forceinline__ unsigned short f2bf(float f) {
  unsigned u = __builtin_bit_cast(unsigned, f);
  u += 0x7FFFu + ((u >> 16) & 1u);          // round-to-nearest-even
  return (unsigned short)(u >> 16);
}

static __device__ __forceinline__ float sigmoidf(float s) {
  return 1.0f / (1.0f + __expf(-s));
}

// low 32 bits of a generic pointer into LDS == wave-relative LDS byte offset
static __device__ __forceinline__ unsigned lds_off(const void* p) {
  return (unsigned)(unsigned long long)p;
}

// GLOBAL_LOAD_ASYNC_TO_LDS_B128, GV mode: 16 bytes/lane, global -> LDS,
// tracked by ASYNCcnt, no VGPR staging.
static __device__ __forceinline__ void async_b128(unsigned dst_lds, unsigned long long src) {
  asm volatile("global_load_async_to_lds_b128 %0, %1, off"
               :: "v"(dst_lds), "v"(src) : "memory");
}

// ---------------------------------------------------------------------------
// Kernel 0: streaming f32 -> bf16 conversion (hoisted out of the GEMM loop).
// ---------------------------------------------------------------------------
__global__ __launch_bounds__(256)
void cvt_bf16_kernel(const float* __restrict__ src, unsigned short* __restrict__ dst) {
  int i = blockIdx.x * blockDim.x + threadIdx.x;   // one float4 per thread
  const float4 f = ((const float4*)src)[i];
  v4us p = { f2bf(f.x), f2bf(f.y), f2bf(f.z), f2bf(f.w) };
  ((v4us*)dst)[i] = p;
}

// ---------------------------------------------------------------------------
// Kernel 1: fused dual GEMM on pre-converted bf16, async-to-LDS double buffer.
//   alpha = sigmoid(x Wa^T + ba) -> d_out[0 .. M*D)      (output slot, temp)
//   v     =         x Wv^T + bv  -> h-region rows 1..T   (h slot, temp)
// Block = 256 threads (8 wave32). Block tile 128(M) x 64(N), K-step 64.
// Each wave: 2x2 WMMA tiles x 2 weight matrices, 16 WMMAs per K-step.
// ---------------------------------------------------------------------------
__global__ __launch_bounds__(256, 2)
void gates_gemm_kernel(const unsigned short* __restrict__ xb,
                       const unsigned short* __restrict__ Wab,
                       const float* __restrict__ ba,
                       const unsigned short* __restrict__ Wvb,
                       const float* __restrict__ bv,
                       float* __restrict__ out) {
  __shared__ unsigned short As [2][128 * LDA];   // A tile (M x K), bf16, padded
  __shared__ unsigned short Bas[2][64 * LDA];    // W_alpha tile (N x K)
  __shared__ unsigned short Bvs[2][64 * LDA];    // W_v tile (N x K)

  const int tid   = threadIdx.x;
  const int lane  = tid & 31;
  const int wave  = tid >> 5;
  const int waveM = wave & 3;                 // 0..3 -> M sub-tile of 32
  const int waveN = wave >> 2;                // 0..1 -> N sub-tile of 32
  const int m0 = blockIdx.y * 128;
  const int n0 = blockIdx.x * 64;

  const int halfsel = lane >> 4;              // 0: lanes 0-15, 1: lanes 16-31
  const int r16     = lane & 15;
  const int kb      = halfsel * 8;            // K-offset per ISA 16-bit A/B layout

  // ---- precompute per-thread async source addresses + LDS destinations ----
  unsigned long long ga[4], gwa[2], gwv[2];
  unsigned lA[2][4], lWa[2][2], lWv[2][2];
#pragma unroll
  for (int it = 0; it < 4; ++it) {
    int id  = tid + it * 256;                 // 0..1023
    int row = id >> 3;                        // 0..127
    int c8  = id & 7;                         // 8-short chunk
    ga[it] = (unsigned long long)&xb[(size_t)(m0 + row) * D_DIM + c8 * 8];
    lA[0][it] = lds_off(&As[0][row * LDA + c8 * 8]);
    lA[1][it] = lds_off(&As[1][row * LDA + c8 * 8]);
  }
#pragma unroll
  for (int it = 0; it < 2; ++it) {
    int id  = tid + it * 256;                 // 0..511
    int row = id >> 3;                        // 0..63
    int c8  = id & 7;
    gwa[it] = (unsigned long long)&Wab[(size_t)(n0 + row) * D_DIM + c8 * 8];
    gwv[it] = (unsigned long long)&Wvb[(size_t)(n0 + row) * D_DIM + c8 * 8];
    lWa[0][it] = lds_off(&Bas[0][row * LDA + c8 * 8]);
    lWa[1][it] = lds_off(&Bas[1][row * LDA + c8 * 8]);
    lWv[0][it] = lds_off(&Bvs[0][row * LDA + c8 * 8]);
    lWv[1][it] = lds_off(&Bvs[1][row * LDA + c8 * 8]);
  }

#define ISSUE_TILE(b)                                                         \
  do {                                                                        \
    _Pragma("unroll") for (int it = 0; it < 4; ++it) {                        \
      async_b128(lA[b][it], ga[it]);  ga[it]  += KSTEP * 2;                   \
    }                                                                         \
    _Pragma("unroll") for (int it = 0; it < 2; ++it) {                        \
      async_b128(lWa[b][it], gwa[it]); gwa[it] += KSTEP * 2;                  \
      async_b128(lWv[b][it], gwv[it]); gwv[it] += KSTEP * 2;                  \
    }                                                                         \
  } while (0)

  v8f acc_a[2][2] = {};
  v8f acc_v[2][2] = {};

  ISSUE_TILE(0);                              // prologue: fetch tile 0 -> buf 0
  int cur = 0;

#pragma unroll 2
  for (int k0 = 0; k0 < D_DIM; k0 += KSTEP) {
    if (k0 + KSTEP < D_DIM) {
      ISSUE_TILE(cur ^ 1);                    // prefetch next tile into other buf
      asm volatile("s_wait_asynccnt 0x8" ::: "memory");  // current tile landed
    } else {
      asm volatile("s_wait_asynccnt 0x0" ::: "memory");
    }
    __syncthreads();

    const unsigned short* Ac  = As[cur];
    const unsigned short* Bac = Bas[cur];
    const unsigned short* Bvc = Bvs[cur];

#pragma unroll
    for (int ks = 0; ks < KSTEP; ks += 32) {
      v16bf afrag[2], bafrag[2], bvfrag[2];
#pragma unroll
      for (int mi = 0; mi < 2; ++mi) {
        int row = waveM * 32 + mi * 16 + r16;
        v8us lo = *(const v8us*)&Ac[row * LDA + ks + kb];
        v8us hi = *(const v8us*)&Ac[row * LDA + ks + kb + 16];
        v16us f = __builtin_shufflevector(lo, hi, 0,1,2,3,4,5,6,7,8,9,10,11,12,13,14,15);
        afrag[mi] = __builtin_bit_cast(v16bf, f);
      }
#pragma unroll
      for (int ni = 0; ni < 2; ++ni) {
        int row = waveN * 32 + ni * 16 + r16;
        {
          v8us lo = *(const v8us*)&Bac[row * LDA + ks + kb];
          v8us hi = *(const v8us*)&Bac[row * LDA + ks + kb + 16];
          v16us f = __builtin_shufflevector(lo, hi, 0,1,2,3,4,5,6,7,8,9,10,11,12,13,14,15);
          bafrag[ni] = __builtin_bit_cast(v16bf, f);
        }
        {
          v8us lo = *(const v8us*)&Bvc[row * LDA + ks + kb];
          v8us hi = *(const v8us*)&Bvc[row * LDA + ks + kb + 16];
          v16us f = __builtin_shufflevector(lo, hi, 0,1,2,3,4,5,6,7,8,9,10,11,12,13,14,15);
          bvfrag[ni] = __builtin_bit_cast(v16bf, f);
        }
      }
#pragma unroll
      for (int mi = 0; mi < 2; ++mi)
#pragma unroll
        for (int ni = 0; ni < 2; ++ni) {
          acc_a[mi][ni] = __builtin_amdgcn_wmma_f32_16x16x32_bf16(
              false, afrag[mi], false, bafrag[ni], (short)0, acc_a[mi][ni], false, false);
          acc_v[mi][ni] = __builtin_amdgcn_wmma_f32_16x16x32_bf16(
              false, afrag[mi], false, bvfrag[ni], (short)0, acc_v[mi][ni], false, false);
        }
    }
    __syncthreads();                          // reads done before buf refilled
    cur ^= 1;
  }
#undef ISSUE_TILE

  // ---- epilogue: bias + sigmoid; scatter per C/D VGPR layout ----
  float* aout = out;                                   // alpha parked in output slot
  float* hout = out + (size_t)M_DIM * D_DIM;           // h region, (T+1)*BD floats

#pragma unroll
  for (int ni = 0; ni < 2; ++ni) {
    int n = n0 + waveN * 32 + ni * 16 + r16;           // N = lane&15 for both halves
    float biasA = ba[n];
    float biasV = bv[n];
#pragma unroll
    for (int mi = 0; mi < 2; ++mi) {
#pragma unroll
      for (int r = 0; r < 8; ++r) {
        int m = m0 + waveM * 32 + mi * 16 + halfsel * 8 + r;   // VGPR r -> M=r / M=r+8
        float alpha = sigmoidf(acc_a[mi][ni][r] + biasA);
        aout[(size_t)m * D_DIM + n] = alpha;
        float v = acc_v[mi][ni][r] + biasV;
        hout[(size_t)(m + B_DIM) * D_DIM + n] = v;     // row m=t*B+b -> h[t+1,b,n]
      }
    }
  }
}

// ---------------------------------------------------------------------------
// Scan pass 1: per-chunk composition. h_end = P*h_start + S.  512K threads.
// ---------------------------------------------------------------------------
__global__ __launch_bounds__(256)
void chunk_compose_kernel(const float* __restrict__ out_base, float* __restrict__ ws) {
  int idx = blockIdx.x * blockDim.x + threadIdx.x;     // 0 .. NCHUNK*BD-1
  int c = idx >> 13;                                   // / BD
  int i = idx & (BD - 1);
  const float* aptr = out_base;                                    // alpha[t]
  const float* vptr = out_base + (size_t)M_DIM * D_DIM + BD;       // v[t] at h[t+1]
  float P = 1.0f, S = 0.0f;
  int t0 = c * CHUNKLEN;
#pragma unroll 4
  for (int t = t0; t < t0 + CHUNKLEN; ++t) {
    float a = aptr[(size_t)t * BD + i];
    float v = vptr[(size_t)t * BD + i];
    S = fmaf(a, S, v);
    P *= a;
  }
  ws[(size_t)c * BD + i] = P;
  ws[(size_t)NCHUNK * BD + (size_t)c * BD + i] = S;
}

// ---------------------------------------------------------------------------
// Scan pass 2: serial scan over 64 chunk states per lane; S[c] <- h at chunk
// start; also writes h0 = 0.
// ---------------------------------------------------------------------------
__global__ __launch_bounds__(256)
void chunk_scan_kernel(float* __restrict__ out_base, float* __restrict__ ws) {
  int i = blockIdx.x * blockDim.x + threadIdx.x;       // 0..BD-1
  float* h0 = out_base + (size_t)M_DIM * D_DIM;        // h row 0
  h0[i] = 0.0f;
  float* P = ws;
  float* S = ws + (size_t)NCHUNK * BD;
  float h = 0.0f;
  for (int c = 0; c < NCHUNK; ++c) {
    size_t off = (size_t)c * BD + i;
    float p = P[off];
    float s = S[off];
    S[off] = h;                                        // h at chunk start
    h = fmaf(p, h, s);
  }
}

// ---------------------------------------------------------------------------
// Scan pass 3: replay each chunk from its start state; overwrite alpha->output
// and v->h in place (read-before-write per location, one owner per element).
// ---------------------------------------------------------------------------
__global__ __launch_bounds__(256)
void chunk_apply_kernel(float* __restrict__ out_base, const float* __restrict__ ws) {
  int idx = blockIdx.x * blockDim.x + threadIdx.x;
  int c = idx >> 13;
  int i = idx & (BD - 1);
  float* aptr  = out_base;                                   // alpha in / output out
  float* vhptr = out_base + (size_t)M_DIM * D_DIM + BD;      // v in / h[t+1] out
  const float* Hs = ws + (size_t)NCHUNK * BD;
  float h = Hs[(size_t)c * BD + i];
  int t0 = c * CHUNKLEN;
#pragma unroll 4
  for (int t = t0; t < t0 + CHUNKLEN; ++t) {
    size_t off = (size_t)t * BD + i;
    float a = aptr[off];
    float v = vhptr[off];
    h = fmaf(a, h, v);
    vhptr[off] = h;
    aptr[off] = h * h * sigmoidf(h);                   // h * silu(h)
  }
}

// ---------------------------------------------------------------------------
extern "C" void kernel_launch(void* const* d_in, const int* in_sizes, int n_in,
                              void* d_out, int out_size, void* d_ws, size_t ws_size,
                              hipStream_t stream) {
  const float* x  = (const float*)d_in[0];
  const float* Wa = (const float*)d_in[1];
  const float* ba = (const float*)d_in[2];
  const float* Wv = (const float*)d_in[3];
  const float* bv = (const float*)d_in[4];
  float* out = (float*)d_out;

  // workspace layout: [P 2MB][S 2MB][xb 32MB][Wab 2MB][Wvb 2MB]  (~40 MB)
  float* wsP = (float*)d_ws;
  float* wsS = wsP + (size_t)NCHUNK * BD;
  unsigned short* xb  = (unsigned short*)(wsS + (size_t)NCHUNK * BD);
  unsigned short* Wab = xb  + (size_t)M_DIM * D_DIM;
  unsigned short* Wvb = Wab + (size_t)D_DIM * D_DIM;

  // pre-convert to bf16 (streaming, float4 per thread)
  cvt_bf16_kernel<<<(M_DIM * D_DIM / 4) / 256, 256, 0, stream>>>(x,  xb);
  cvt_bf16_kernel<<<(D_DIM * D_DIM / 4) / 256, 256, 0, stream>>>(Wa, Wab);
  cvt_bf16_kernel<<<(D_DIM * D_DIM / 4) / 256, 256, 0, stream>>>(Wv, Wvb);

  dim3 gemm_grid(D_DIM / 64, M_DIM / 128);             // (16, 128)
  gates_gemm_kernel<<<gemm_grid, 256, 0, stream>>>(xb, Wab, ba, Wvb, bv, out);

  chunk_compose_kernel<<<(NCHUNK * BD) / 256, 256, 0, stream>>>(out, (float*)d_ws);
  chunk_scan_kernel<<<BD / 256, 256, 0, stream>>>(out, (float*)d_ws);
  chunk_apply_kernel<<<(NCHUNK * BD) / 256, 256, 0, stream>>>(out, (float*)d_ws);
}
